// GraphCON_GAT_18107582120778
// MI455X (gfx1250) — compile-verified, matrix operand order checked by reading, not run on hardware
//
#include <hip/hip_runtime.h>
#include <hip/hip_bf16.h>
#include <math.h>

#define NNODES 10000
#define NFEAT  256
#define NHID   128
#define NHEAD  4
#define NCLASS 40
#define HDIM   (NHEAD * NHID)   // 512
#define NEG_SLOPE 0.2f
#define DT_C    1.0f
#define ALPHA_C 1.0f
#define GAMMA_C 1.0f

typedef _Float16 v8h  __attribute__((ext_vector_type(8)));
typedef _Float16 v16h __attribute__((ext_vector_type(16)));
typedef float    v8f  __attribute__((ext_vector_type(8)));

// ---------------------------------------------------------------------------
// f32 -> f16 conversion kernels
// ---------------------------------------------------------------------------
__global__ void cvt_f16_kernel(const float* __restrict__ s, _Float16* __restrict__ d, int n) {
    int i = blockIdx.x * blockDim.x + threadIdx.x;
    if (i < n) d[i] = (_Float16)s[i];
}

__global__ void cvt_f16_pad_kernel(const float* __restrict__ s, _Float16* __restrict__ d,
                                   int rows, int cols, int padRows) {
    int i = blockIdx.x * blockDim.x + threadIdx.x;
    if (i >= padRows * cols) return;
    int r = i / cols;
    d[i] = (r < rows) ? (_Float16)s[i] : (_Float16)0.f;
}

// ---------------------------------------------------------------------------
// WMMA GEMM: out[M,Nreal] = act(A[M,K] @ W[Npad,K]^T + bias)
// One wave computes a 16 x (16*NT) output strip: the A fragment is loaded
// once per k-step and reused across NT independent WMMA accumulator chains
// (cuts A re-reads NT x and hides WMMA->WMMA RAW latency).
// Operand layout per CDNA5 ISA 7.12.2 (16-bit A/B, 16x16x32):
//   lanes 0-15 : row = lane,    elems 0..7 = K 0..7,   8..15 = K 16..23
//   lanes16-31 : row = lane-16, elems 0..7 = K 8..15,  8..15 = K 24..31
// act: 0 = none, 1 = relu
// ---------------------------------------------------------------------------
template<int NT>
__global__ void wmma_gemm_kernel(const _Float16* __restrict__ A,
                                 const _Float16* __restrict__ W,
                                 const float* __restrict__ bias,
                                 float* __restrict__ out,
                                 int M, int K, int Nreal, int act)
{
    const int lane = threadIdx.x & 31;
    const int tm = blockIdx.x * 16;
    const int tn = blockIdx.y * (16 * NT);
    const int rA = tm + (lane & 15);
    const int kh = (lane >> 4) << 3;          // 0 or 8
    const _Float16* pa = A + (size_t)rA * K + kh;
    const _Float16* pb[NT];
#pragma unroll
    for (int t = 0; t < NT; ++t)
        pb[t] = W + (size_t)(tn + t * 16 + (lane & 15)) * K + kh;

    v8f acc[NT];
#pragma unroll
    for (int t = 0; t < NT; ++t) acc[t] = (v8f){};

    for (int k0 = 0; k0 < K; k0 += 32) {
        v8h alo = *(const v8h*)(pa + k0);
        v8h ahi = *(const v8h*)(pa + k0 + 16);
        v16h a = __builtin_shufflevector(alo, ahi,
                     0,1,2,3,4,5,6,7,8,9,10,11,12,13,14,15);
#pragma unroll
        for (int t = 0; t < NT; ++t) {
            v8h blo = *(const v8h*)(pb[t] + k0);
            v8h bhi = *(const v8h*)(pb[t] + k0 + 16);
            v16h b = __builtin_shufflevector(blo, bhi,
                         0,1,2,3,4,5,6,7,8,9,10,11,12,13,14,15);
            acc[t] = __builtin_amdgcn_wmma_f32_16x16x32_f16(
                /*neg_a=*/false, a, /*neg_b=*/false, b,
                /*c_mod=*/(short)0, acc[t], /*reuse_a=*/false, /*reuse_b=*/false);
        }
    }

    // C/D layout: lane&15 = N col; VGPR v -> M row v (+8 for lanes 16-31)
    const int mb = tm + ((lane >> 4) << 3);
#pragma unroll
    for (int t = 0; t < NT; ++t) {
        const int n = tn + t * 16 + (lane & 15);
        if (n < Nreal) {
            const float bv = bias ? bias[n] : 0.f;
#pragma unroll
            for (int v = 0; v < 8; ++v) {
                float val = acc[t][v] + bv;
                if (act == 1) val = fmaxf(val, 0.f);
                out[(size_t)(mb + v) * Nreal + n] = val;
            }
        }
    }
}

// ---------------------------------------------------------------------------
// Attention scores: a_src[n,h] = dot(xh[n,h,:], att_src[h,:]); same for dst.
// One wave per (n,h); 4 channels/lane; wave reduction.
// ---------------------------------------------------------------------------
__global__ void att_kernel(const float* __restrict__ xh,
                           const float* __restrict__ att_src,
                           const float* __restrict__ att_dst,
                           float* __restrict__ a_src, float* __restrict__ a_dst)
{
    const int wave = threadIdx.x >> 5;
    const int lane = threadIdx.x & 31;
    const int idx = blockIdx.x * 8 + wave;     // (n,h) flat
    if (idx >= NNODES * NHEAD) return;
    const int n = idx >> 2, h = idx & 3;
    const float4 xv  = *(const float4*)(xh + (size_t)n * HDIM + h * NHID + lane * 4);
    const float4 asv = *(const float4*)(att_src + h * NHID + lane * 4);
    const float4 adv = *(const float4*)(att_dst + h * NHID + lane * 4);
    float ss = xv.x*asv.x + xv.y*asv.y + xv.z*asv.z + xv.w*asv.w;
    float sd = xv.x*adv.x + xv.y*adv.y + xv.z*adv.z + xv.w*adv.w;
#pragma unroll
    for (int off = 16; off > 0; off >>= 1) {
        ss += __shfl_xor(ss, off, 32);
        sd += __shfl_xor(sd, off, 32);
    }
    if (lane == 0) { a_src[idx] = ss; a_dst[idx] = sd; }
}

// ---------------------------------------------------------------------------
// Edge helpers
// ---------------------------------------------------------------------------
__device__ __forceinline__ void edge_sd(const int* __restrict__ ei, int E, int idx,
                                        int& s, int& d) {
    if (idx < E) { s = ei[idx]; d = ei[E + idx]; }
    else         { s = idx - E; d = idx - E; }       // self-loops
}
__device__ __forceinline__ float lrelu(float x) { return x > 0.f ? x : NEG_SLOPE * x; }
// order-preserving float -> uint key (for atomicMax-based segment max)
__device__ __forceinline__ unsigned f2key(float f) {
    unsigned b = __float_as_uint(f);
    return (b & 0x80000000u) ? ~b : (b | 0x80000000u);
}
__device__ __forceinline__ float key2f(unsigned k) {
    unsigned b = (k & 0x80000000u) ? (k & 0x7FFFFFFFu) : ~k;
    return __uint_as_float(b);
}

__global__ void edge_max_kernel(const int* __restrict__ ei, int E, int Etot,
                                const float* __restrict__ a_src,
                                const float* __restrict__ a_dst,
                                unsigned* __restrict__ mkeys)
{
    int t = blockIdx.x * blockDim.x + threadIdx.x;
    if (t >= Etot * NHEAD) return;
    int idx = t >> 2, h = t & 3;
    int s, d; edge_sd(ei, E, idx, s, d);
    float e = lrelu(a_src[s * NHEAD + h] + a_dst[d * NHEAD + h]);
    atomicMax(&mkeys[d * NHEAD + h], f2key(e));
}

__global__ void edge_exp_kernel(const int* __restrict__ ei, int E, int Etot,
                                const float* __restrict__ a_src,
                                const float* __restrict__ a_dst,
                                const unsigned* __restrict__ mkeys,
                                float* __restrict__ ew, float* __restrict__ ssum)
{
    int t = blockIdx.x * blockDim.x + threadIdx.x;
    if (t >= Etot * NHEAD) return;
    int idx = t >> 2, h = t & 3;
    int s, d; edge_sd(ei, E, idx, s, d);
    float e = lrelu(a_src[s * NHEAD + h] + a_dst[d * NHEAD + h]);
    float m = key2f(mkeys[d * NHEAD + h]);
    float w = __expf(e - m);
    ew[(size_t)idx * NHEAD + h] = w;
    unsafeAtomicAdd(&ssum[d * NHEAD + h], w);
}

// Wave per edge: scatter msg = xh[src] * alpha into conv[dst] (512 floats)
__global__ void edge_scatter_kernel(const int* __restrict__ ei, int E, int Etot,
                                    const float* __restrict__ xh,
                                    const float* __restrict__ ew,
                                    const float* __restrict__ ssum,
                                    float* __restrict__ conv)
{
    const int wave = threadIdx.x >> 5;
    const int lane = threadIdx.x & 31;
    const int idx = blockIdx.x * 8 + wave;
    if (idx >= Etot) return;
    int s, d; edge_sd(ei, E, idx, s, d);
    float al[NHEAD];
#pragma unroll
    for (int h = 0; h < NHEAD; ++h)
        al[h] = ew[(size_t)idx * NHEAD + h] / ssum[d * NHEAD + h];
    const float* xs = xh + (size_t)s * HDIM;
    float* cd = conv + (size_t)d * HDIM;
#pragma unroll
    for (int t = 0; t < 16; ++t) {
        int j = lane + (t << 5);
        unsafeAtomicAdd(&cd[j], xs[j] * al[j >> 7]);
    }
}

// ---------------------------------------------------------------------------
// z = mean_h elu(conv + gat_b + res) over flat groups of 4 (view(N,NHID,H));
// GraphCON update: Y += DT*(z - ALPHA*Y - GAMMA*X); X += DT*Y  (in place)
// ---------------------------------------------------------------------------
__global__ void combine_kernel(const float* __restrict__ conv,
                               const float* __restrict__ res,
                               const float* __restrict__ gat_b,
                               float* __restrict__ Y, float* __restrict__ X)
{
    int i = blockIdx.x * blockDim.x + threadIdx.x;    // (n, c)
    if (i >= NNODES * NHID) return;
    int n = i >> 7, c = i & 127;
    const float4 cv = *(const float4*)(conv + (size_t)n * HDIM + c * 4);
    const float4 rv = *(const float4*)(res  + (size_t)n * HDIM + c * 4);
    const float4 gb = *(const float4*)(gat_b + c * 4);
    float t0 = cv.x + gb.x + rv.x; t0 = t0 > 0.f ? t0 : __expf(t0) - 1.f;
    float t1 = cv.y + gb.y + rv.y; t1 = t1 > 0.f ? t1 : __expf(t1) - 1.f;
    float t2 = cv.z + gb.z + rv.z; t2 = t2 > 0.f ? t2 : __expf(t2) - 1.f;
    float t3 = cv.w + gb.w + rv.w; t3 = t3 > 0.f ? t3 : __expf(t3) - 1.f;
    float z = 0.25f * (t0 + t1 + t2 + t3);
    float y = Y[i], x = X[i];
    float yn = y + DT_C * (z - ALPHA_C * y - GAMMA_C * x);
    float xn = x + DT_C * yn;
    Y[i] = yn; X[i] = xn;
}

// ---------------------------------------------------------------------------
// Host-side orchestration
// ---------------------------------------------------------------------------
extern "C" void kernel_launch(void* const* d_in, const int* in_sizes, int n_in,
                              void* d_out, int out_size, void* d_ws, size_t ws_size,
                              hipStream_t stream)
{
    const float* x     = (const float*)d_in[0];
    const int*   ei    = (const int*)  d_in[1];
    const float* enc_w = (const float*)d_in[2];
    const float* enc_b = (const float*)d_in[3];
    const float* res_w = (const float*)d_in[4];
    const float* res_b = (const float*)d_in[5];
    const float* gat_w = (const float*)d_in[6];
    const float* att_s = (const float*)d_in[7];
    const float* att_d = (const float*)d_in[8];
    const float* gat_b = (const float*)d_in[9];
    const float* dec_w = (const float*)d_in[10];
    const float* dec_b = (const float*)d_in[11];
    float* out = (float*)d_out;

    const int E    = in_sizes[1] / 2;
    const int Etot = E + NNODES;

    // bump allocator over workspace
    char* p = (char*)d_ws;
    auto alloc = [&](size_t b) -> void* {
        void* r = (void*)p; p += (b + 255) & ~(size_t)255; return r;
    };
    _Float16* act16   = (_Float16*)alloc((size_t)NNODES * NFEAT * 2);
    _Float16* enc_w16 = (_Float16*)alloc((size_t)NHID * NFEAT * 2);
    _Float16* gat_w16 = (_Float16*)alloc((size_t)HDIM * NHID * 2);
    _Float16* res_w16 = (_Float16*)alloc((size_t)HDIM * NHID * 2);
    _Float16* dec_w16 = (_Float16*)alloc((size_t)48 * NHID * 2);
    float* Y     = (float*)alloc((size_t)NNODES * NHID * 4);
    float* X     = (float*)alloc((size_t)NNODES * NHID * 4);
    float* xh    = (float*)alloc((size_t)NNODES * HDIM * 4);
    float* resb  = (float*)alloc((size_t)NNODES * HDIM * 4);
    float* conv  = (float*)alloc((size_t)NNODES * HDIM * 4);
    float* a_src = (float*)alloc((size_t)NNODES * NHEAD * 4);
    float* a_dst = (float*)alloc((size_t)NNODES * NHEAD * 4);
    unsigned* mkeys = (unsigned*)alloc((size_t)NNODES * NHEAD * 4);
    float* ssum  = (float*)alloc((size_t)NNODES * NHEAD * 4);
    float* ew    = (float*)alloc((size_t)Etot * NHEAD * 4);

    const dim3 b256(256);
    auto cdiv = [](int a, int b) { return (a + b - 1) / b; };

    // Convert weights to f16 (zero-pad decoder rows 40..47)
    cvt_f16_kernel<<<cdiv(NHID*NFEAT,256), b256, 0, stream>>>(enc_w, enc_w16, NHID*NFEAT);
    cvt_f16_kernel<<<cdiv(HDIM*NHID,256),  b256, 0, stream>>>(gat_w, gat_w16, HDIM*NHID);
    cvt_f16_kernel<<<cdiv(HDIM*NHID,256),  b256, 0, stream>>>(res_w, res_w16, HDIM*NHID);
    cvt_f16_pad_kernel<<<cdiv(48*NHID,256), b256, 0, stream>>>(dec_w, dec_w16, NCLASS, NHID, 48);

    // Encoder: Y = relu(x @ enc_w^T + enc_b); X = Y
    cvt_f16_kernel<<<cdiv(NNODES*NFEAT,256), b256, 0, stream>>>(x, act16, NNODES*NFEAT);
    wmma_gemm_kernel<4><<<dim3(NNODES/16, NHID/64), dim3(32), 0, stream>>>(
        act16, enc_w16, enc_b, Y, NNODES, NFEAT, NHID, /*relu*/1);
    hipMemcpyAsync(X, Y, (size_t)NNODES * NHID * 4, hipMemcpyDeviceToDevice, stream);

    for (int layer = 0; layer < 2; ++layer) {
        cvt_f16_kernel<<<cdiv(NNODES*NHID,256), b256, 0, stream>>>(X, act16, NNODES*NHID);
        wmma_gemm_kernel<4><<<dim3(NNODES/16, HDIM/64), dim3(32), 0, stream>>>(
            act16, gat_w16, nullptr, xh, NNODES, NHID, HDIM, 0);
        wmma_gemm_kernel<4><<<dim3(NNODES/16, HDIM/64), dim3(32), 0, stream>>>(
            act16, res_w16, res_b, resb, NNODES, NHID, HDIM, 0);
        att_kernel<<<cdiv(NNODES*NHEAD,8), b256, 0, stream>>>(xh, att_s, att_d, a_src, a_dst);
        hipMemsetAsync(mkeys, 0, (size_t)NNODES * NHEAD * 4, stream);
        hipMemsetAsync(ssum,  0, (size_t)NNODES * NHEAD * 4, stream);
        hipMemsetAsync(conv,  0, (size_t)NNODES * HDIM * 4, stream);
        edge_max_kernel<<<cdiv(Etot*NHEAD,256), b256, 0, stream>>>(ei, E, Etot, a_src, a_dst, mkeys);
        edge_exp_kernel<<<cdiv(Etot*NHEAD,256), b256, 0, stream>>>(ei, E, Etot, a_src, a_dst, mkeys, ew, ssum);
        edge_scatter_kernel<<<cdiv(Etot,8), b256, 0, stream>>>(ei, E, Etot, xh, ew, ssum, conv);
        combine_kernel<<<cdiv(NNODES*NHID,256), b256, 0, stream>>>(conv, resb, gat_b, Y, X);
    }

    // Decoder: out = X @ dec_w^T + dec_b  (N padded 40 -> 48, NT=3)
    cvt_f16_kernel<<<cdiv(NNODES*NHID,256), b256, 0, stream>>>(X, act16, NNODES*NHID);
    wmma_gemm_kernel<3><<<dim3(NNODES/16, 1), dim3(32), 0, stream>>>(
        act16, dec_w16, dec_b, out, NNODES, NHID, NCLASS, 0);
}
